// GQA_82635170775076
// MI455X (gfx1250) — matrix operand order repaired and checked by reading, not compiled, not run on hardware
//
#include <hip/hip_runtime.h>

typedef __attribute__((ext_vector_type(16))) __bf16 v16bf;
typedef __attribute__((ext_vector_type(8)))  float  v8f;

union FragBF {
  v16bf  v;
  uint4  u4[2];
  __bf16 h[16];
};

union PackBF4 {
  __bf16 e[4];
  uint2  u;
};

#define TSEQ 2048
#define HID  4096
#define NQH  32
#define NKVH 8
#define DH   128

// Async bf16 tile copy: global -> LDS, no VGPR round-trip (ASYNCcnt-tracked).
__device__ __forceinline__ void async_copy_b128(void* lds_dst, const void* gsrc) {
  unsigned lds_off = (unsigned)(uintptr_t)lds_dst;   // low 32 bits = LDS byte offset
  asm volatile("global_load_async_to_lds_b128 %0, %1, off"
               :: "v"(lds_off), "v"(gsrc) : "memory");
}
__device__ __forceinline__ void wait_asynccnt0() {
  asm volatile("s_wait_asynccnt 0" ::: "memory");
}

// ------------------------------------------------------------------
// Tiled GEMM: C = A * B, A is MxK (fp32 or bf16), B is KxN fp32.
// bf16 WMMA, f32 accumulation, double-buffered LDS.
// Block = 128x128 tile, 256 threads = 8 waves; wave = 32x64 micro-tile.
// bf16-A path stages A via GLOBAL_LOAD_ASYNC_TO_LDS_B128.
// ------------------------------------------------------------------
#define BM 128
#define BN 128
#define BK 32
#define LDT (BK + 8)   // 40 bf16 = 80B row pitch (16B aligned)

template <typename AT, bool OUT_BF16>
__global__ __launch_bounds__(256)
void gemm_wmma(const AT* __restrict__ A, const float* __restrict__ B,
               void* __restrict__ Cout, int M, int N, int K)
{
  __shared__ __bf16 As[2][BM][LDT];   // [buf][m][k]
  __shared__ __bf16 Bs[2][BN][LDT];   // [buf][n][k]  (B transposed into LDS)

  const int tid  = threadIdx.x;
  const int bm   = blockIdx.x * BM;
  const int bn   = blockIdx.y * BN;
  const int wave = tid >> 5;
  const int lane = tid & 31;
  const int ln16 = lane & 15;
  const int hl   = lane >> 4;      // lane half: 0 / 1
  const int kA   = hl * 8;         // A-frag K base (ISA 16-bit A layout)
  const int kB   = hl * 16;        // unused; B frag uses kA (same k order as A)
  const int wm   = (wave & 3) * 32;
  const int wn   = (wave >> 2) * 64;

  const v8f z8 = {0.f,0.f,0.f,0.f,0.f,0.f,0.f,0.f};
  v8f acc[2][4];
#pragma unroll
  for (int i = 0; i < 2; ++i)
#pragma unroll
    for (int j = 0; j < 4; ++j) acc[i][j] = z8;

  auto stage = [&](int sb, int k0) {
    // --- A tile (128x32) ---
    if constexpr (sizeof(AT) == 4) {
      // fp32 -> bf16 conversion, packed 8-byte LDS stores
#pragma unroll
      for (int i = 0; i < 4; ++i) {
        int s  = tid + i * 256;          // 0..1023
        int r  = s >> 3;
        int c4 = (s & 7) * 4;
        float4 f = *reinterpret_cast<const float4*>(
            reinterpret_cast<const float*>(A) + (size_t)(bm + r) * K + k0 + c4);
        PackBF4 pk;
        pk.e[0] = (__bf16)f.x; pk.e[1] = (__bf16)f.y;
        pk.e[2] = (__bf16)f.z; pk.e[3] = (__bf16)f.w;
        *reinterpret_cast<uint2*>(&As[sb][r][c4]) = pk.u;
      }
    } else {
      // already bf16: async DMA global -> LDS (ASYNCcnt)
#pragma unroll
      for (int i = 0; i < 2; ++i) {
        int s  = tid + i * 256;          // 0..511
        int r  = s >> 2;
        int c8 = (s & 3) * 8;
        async_copy_b128(&As[sb][r][c8],
            reinterpret_cast<const __bf16*>(A) + (size_t)(bm + r) * K + k0 + c8);
      }
    }
    // --- B tile (32x128) -> bf16, transposed (scalar stores) ---
#pragma unroll
    for (int i = 0; i < 4; ++i) {
      int s  = tid + i * 256;            // 0..1023
      int kr = s >> 5;
      int c4 = (s & 31) * 4;
      float4 f = *reinterpret_cast<const float4*>(B + (size_t)(k0 + kr) * N + bn + c4);
      Bs[sb][c4+0][kr] = (__bf16)f.x; Bs[sb][c4+1][kr] = (__bf16)f.y;
      Bs[sb][c4+2][kr] = (__bf16)f.z; Bs[sb][c4+3][kr] = (__bf16)f.w;
    }
  };

  const int nk = K / BK;
  stage(0, 0);
  int buf = 0;

  for (int kt = 0; kt < nk; ++kt) {
    if constexpr (sizeof(AT) == 2) wait_asynccnt0();  // own async writes done
    __syncthreads();                                  // publish tile to block
    if (kt + 1 < nk) stage(buf ^ 1, (kt + 1) * BK);
    if (kt + 2 < nk) {
      // pull the B stream into GL2 ahead of time (global_prefetch_b8)
      __builtin_prefetch(B + (size_t)((kt + 2) * BK + (tid >> 3)) * N + bn + (tid & 7) * 16, 0, 1);
    }

    FragBF fa[2], fb[4];
#pragma unroll
    for (int i = 0; i < 2; ++i) {
      const int ar = wm + i * 16 + ln16;
      fa[i].u4[0] = *reinterpret_cast<const uint4*>(&As[buf][ar][kA]);
      fa[i].u4[1] = *reinterpret_cast<const uint4*>(&As[buf][ar][16 + kA]);
    }
#pragma unroll
    for (int j = 0; j < 4; ++j) {
      const int bc = wn + j * 16 + ln16;
      fb[j].u4[0] = *reinterpret_cast<const uint4*>(&Bs[buf][bc][kA]);
      fb[j].u4[1] = *reinterpret_cast<const uint4*>(&Bs[buf][bc][16 + kA]);
    }
#pragma unroll
    for (int i = 0; i < 2; ++i)
#pragma unroll
      for (int j = 0; j < 4; ++j)
        acc[i][j] = __builtin_amdgcn_wmma_f32_16x16x32_bf16(
            false, fa[i].v, false, fb[j].v, (short)0, acc[i][j], false, false);
    buf ^= 1;
  }

  const int r0 = bm + wm + hl * 8;
  const int c0 = bn + wn + ln16;
  if constexpr (OUT_BF16) {
    __bf16* Cb = reinterpret_cast<__bf16*>(Cout);
#pragma unroll
    for (int i = 0; i < 2; ++i)
#pragma unroll
      for (int j = 0; j < 4; ++j)
#pragma unroll
        for (int r = 0; r < 8; ++r)
          Cb[(size_t)(r0 + i*16 + r) * N + c0 + j*16] = (__bf16)acc[i][j][r];
  } else {
    float* Cf = reinterpret_cast<float*>(Cout);
#pragma unroll
    for (int i = 0; i < 2; ++i)
#pragma unroll
      for (int j = 0; j < 4; ++j)
#pragma unroll
        for (int r = 0; r < 8; ++r)
          Cf[(size_t)(r0 + i*16 + r) * N + c0 + j*16] = acc[i][j][r];
  }
}

// ------------------------------------------------------------------
// RoPE on bf16 Q (scaled by 1/sqrt(D)) and K. One block per timestep.
// ------------------------------------------------------------------
__global__ __launch_bounds__(256)
void rope_kernel(__bf16* __restrict__ qb, __bf16* __restrict__ kb)
{
  const int t = blockIdx.x;
  const float qscale = 0.08838834764831845f;   // 1/sqrt(128)
#pragma unroll
  for (int it = 0; it < NQH * (DH/2) / 256; ++it) {
    int p = threadIdx.x + it * 256;
    int head = p >> 6, i = p & 63;
    float ang = (float)t * __powf(10000.0f, -(float)i * (1.0f / 64.0f));
    float c = __cosf(ang), s = __sinf(ang);
    size_t base = (size_t)t * (NQH*DH) + head * DH + i;
    float a = (float)qb[base], b = (float)qb[base + 64];
    qb[base]      = (__bf16)((a * c - b * s) * qscale);
    qb[base + 64] = (__bf16)((b * c + a * s) * qscale);
  }
#pragma unroll
  for (int it = 0; it < NKVH * (DH/2) / 256; ++it) {
    int p = threadIdx.x + it * 256;
    int head = p >> 6, i = p & 63;
    float ang = (float)t * __powf(10000.0f, -(float)i * (1.0f / 64.0f));
    float c = __cosf(ang), s = __sinf(ang);
    size_t base = (size_t)t * (NKVH*DH) + head * DH + i;
    float a = (float)kb[base], b = (float)kb[base + 64];
    kb[base]      = (__bf16)(a * c - b * s);
    kb[base + 64] = (__bf16)(b * c + a * s);
  }
}

// ------------------------------------------------------------------
// Flash attention (causal, GQA). Block = 1 head x 128 query rows,
// 256 threads = 8 waves, each wave owns a 16-row query stripe.
// KV blocks of 32; K tile staged via async-to-LDS DMA.
// ------------------------------------------------------------------
#define FA_BQ  128
#define FA_BKV 32
#define LDK (DH + 8)      // 136 bf16 = 272B pitch (16B aligned)
#define LDP (FA_BKV + 8)  // 40 bf16 = 80B pitch

__global__ __launch_bounds__(256)
void flash_attn_kernel(const __bf16* __restrict__ qb, const __bf16* __restrict__ kb,
                       const __bf16* __restrict__ vb, __bf16* __restrict__ ob)
{
  __shared__ __bf16 Ks[FA_BKV][LDK];   // [kv][d]
  __shared__ __bf16 Vt[DH][LDP];       // [d][kv] (transposed)
  __shared__ __bf16 Ps[8][16][LDP];    // per-wave P stripe [row][kv]

  const int h    = blockIdx.y;
  const int hk   = h >> 2;             // GQA: kv head = h / 4
  const int q0   = blockIdx.x * FA_BQ;
  const int tid  = threadIdx.x;
  const int wave = tid >> 5;
  const int lane = tid & 31;
  const int ln16 = lane & 15;
  const int hl   = lane >> 4;
  const int kA   = hl * 8;
  const int kB   = hl * 16;

  // Q fragments for this wave's 16 rows, 4 K-chunks over D=128
  FragBF qf[4];
  {
    const __bf16* qrow = qb + (size_t)(q0 + wave * 16 + ln16) * (NQH*DH) + h * DH;
#pragma unroll
    for (int c = 0; c < 4; ++c) {
      qf[c].u4[0] = *reinterpret_cast<const uint4*>(qrow + c*32 + kA);
      qf[c].u4[1] = *reinterpret_cast<const uint4*>(qrow + c*32 + 16 + kA);
    }
  }

  const v8f z8 = {0.f,0.f,0.f,0.f,0.f,0.f,0.f,0.f};
  v8f oacc[8];
#pragma unroll
  for (int j = 0; j < 8; ++j) oacc[j] = z8;
  float mrow[8], lrow[8];
#pragma unroll
  for (int r = 0; r < 8; ++r) { mrow[r] = -__builtin_inff(); lrow[r] = 0.f; }

  const int myrow0 = q0 + wave * 16 + hl * 8;

  for (int kvb = 0; kvb < q0 + FA_BQ; kvb += FA_BKV) {
    // ---- stage K via async DMA, V via VGPR + transpose ----
#pragma unroll
    for (int i = 0; i < 2; ++i) {
      int s  = tid + i * 256;          // 0..511
      int r  = s >> 4;                 // kv row 0..31
      int c8 = (s & 15) * 8;           // d 0..120
      async_copy_b128(&Ks[r][c8],
          kb + (size_t)(kvb + r) * (NKVH*DH) + hk * DH + c8);
      union { uint4 u; __bf16 e[8]; } vv;
      vv.u = *reinterpret_cast<const uint4*>(vb + (size_t)(kvb + r) * (NKVH*DH) + hk * DH + c8);
#pragma unroll
      for (int e = 0; e < 8; ++e) Vt[c8 + e][r] = vv.e[e];
    }
    wait_asynccnt0();
    __syncthreads();

    // ---- S = Q * K^T  (16x32 per wave, 2 col-tiles x 4 K-chunks) ----
    v8f sacc[2] = { z8, z8 };
#pragma unroll
    for (int c = 0; c < 4; ++c) {
#pragma unroll
      for (int j = 0; j < 2; ++j) {
        FragBF kf;
        const int kvn = j * 16 + ln16;
        kf.u4[0] = *reinterpret_cast<const uint4*>(&Ks[kvn][c*32 + kA]);
        kf.u4[1] = *reinterpret_cast<const uint4*>(&Ks[kvn][c*32 + 16 + kA]);
        sacc[j] = __builtin_amdgcn_wmma_f32_16x16x32_bf16(
            false, qf[c].v, false, kf.v, (short)0, sacc[j], false, false);
      }
    }

    // ---- online softmax update (per-row over 16-lane halves) ----
#pragma unroll
    for (int r = 0; r < 8; ++r) {
      const int row = myrow0 + r;
      float s0 = sacc[0][r], s1 = sacc[1][r];
      if (kvb + ln16      > row) s0 = -1e30f;
      if (kvb + 16 + ln16 > row) s1 = -1e30f;
      float mx = fmaxf(s0, s1);
#pragma unroll
      for (int off = 8; off >= 1; off >>= 1) mx = fmaxf(mx, __shfl_xor(mx, off, 16));
      const float mnew  = fmaxf(mrow[r], mx);
      const float corr  = __expf(mrow[r] - mnew);
      const float p0    = __expf(s0 - mnew);
      const float p1    = __expf(s1 - mnew);
      float ps = p0 + p1;
#pragma unroll
      for (int off = 8; off >= 1; off >>= 1) ps += __shfl_xor(ps, off, 16);
      lrow[r] = lrow[r] * corr + ps;
      mrow[r] = mnew;
#pragma unroll
      for (int t2 = 0; t2 < 8; ++t2) oacc[t2][r] *= corr;
      Ps[wave][hl*8 + r][ln16]      = (__bf16)p0;
      Ps[wave][hl*8 + r][16 + ln16] = (__bf16)p1;
    }
    __syncthreads();

    // ---- O += P * V ----
    FragBF pf;
    pf.u4[0] = *reinterpret_cast<const uint4*>(&Ps[wave][ln16][kA]);
    pf.u4[1] = *reinterpret_cast<const uint4*>(&Ps[wave][ln16][16 + kA]);
#pragma unroll
    for (int j = 0; j < 8; ++j) {
      FragBF vf;
      const int dn = j * 16 + ln16;
      vf.u4[0] = *reinterpret_cast<const uint4*>(&Vt[dn][kA]);
      vf.u4[1] = *reinterpret_cast<const uint4*>(&Vt[dn][16 + kA]);
      oacc[j] = __builtin_amdgcn_wmma_f32_16x16x32_bf16(
          false, pf.v, false, vf.v, (short)0, oacc[j], false, false);
    }
    __syncthreads();
  }

  // ---- normalize and store bf16 attention output [T, H*D] ----
#pragma unroll
  for (int j = 0; j < 8; ++j)
#pragma unroll
    for (int r = 0; r < 8; ++r) {
      float val = oacc[j][r] / lrow[r];
      ob[(size_t)(myrow0 + r) * (NQH*DH) + h * DH + j*16 + ln16] = (__bf16)val;
    }
}

// ------------------------------------------------------------------
extern "C" void kernel_launch(void* const* d_in, const int* in_sizes, int n_in,
                              void* d_out, int out_size, void* d_ws, size_t ws_size,
                              hipStream_t stream)
{
  (void)in_sizes; (void)n_in; (void)out_size; (void)ws_size;
  const float* x  = (const float*)d_in[0];
  const float* wq = (const float*)d_in[1];
  const float* wk = (const float*)d_in[2];
  const float* wv = (const float*)d_in[3];
  const float* wo = (const float*)d_in[4];
  // d_in[5] = mask (causal handled analytically)

  __bf16* qbuf = (__bf16*)d_ws;                          // 2048*4096
  __bf16* kbuf = qbuf + (size_t)TSEQ * (NQH*DH);         // 2048*1024
  __bf16* vbuf = kbuf + (size_t)TSEQ * (NKVH*DH);        // 2048*1024
  __bf16* obuf = vbuf + (size_t)TSEQ * (NKVH*DH);        // 2048*4096

  dim3 blk(256);
  gemm_wmma<float, true><<<dim3(TSEQ/BM, (NQH*DH)/BN),  blk, 0, stream>>>(x, wq, qbuf, TSEQ, NQH*DH,  HID);
  gemm_wmma<float, true><<<dim3(TSEQ/BM, (NKVH*DH)/BN), blk, 0, stream>>>(x, wk, kbuf, TSEQ, NKVH*DH, HID);
  gemm_wmma<float, true><<<dim3(TSEQ/BM, (NKVH*DH)/BN), blk, 0, stream>>>(x, wv, vbuf, TSEQ, NKVH*DH, HID);

  rope_kernel<<<TSEQ, 256, 0, stream>>>(qbuf, kbuf);

  flash_attn_kernel<<<dim3(TSEQ/FA_BQ, NQH), 256, 0, stream>>>(qbuf, kbuf, vbuf, obuf);

  gemm_wmma<__bf16, false><<<dim3(TSEQ/BM, HID/BN), blk, 0, stream>>>(obuf, wo, d_out, TSEQ, HID, NQH*DH);
}
